// CrossAttentionFusion_29652454211701
// MI455X (gfx1250) — matrix operand order repaired and checked by reading, not compiled, not run on hardware
//
#include <hip/hip_runtime.h>
#include <hip/hip_bf16.h>

typedef _Float16 f16;
typedef __attribute__((ext_vector_type(16))) _Float16 v16h;
typedef __attribute__((ext_vector_type(8)))  _Float16 h8;
typedef __attribute__((ext_vector_type(8)))  float    v8f;

union AF { v16h v; h8 h[2]; };

__device__ __forceinline__ v8f wmma16(const AF& a, const AF& b, v8f c) {
    return __builtin_amdgcn_wmma_f32_16x16x32_f16(
        /*neg_a=*/false, a.v, /*neg_b=*/false, b.v,
        /*c_mod=*/(short)0, c, /*reuse_a=*/false, /*reuse_b=*/false);
}

#define NPOS 4096        // 16*16*16
#define CCH  128
#define HEADS 4
#define HD   32

// ---------------------------------------------------------------------------
// Kernel 1: channel projection  Y[b][n][o] = f16( (sum_c W[o,c] X[b,c,n] + bias[o]) * outScale )
// block: 256 thr (8 waves); covers 128 out-channels x 128 positions.
// grid: B * (NPOS/128) = 64
// ---------------------------------------------------------------------------
__global__ __launch_bounds__(256) void proj_kernel(
    const float* __restrict__ X, const float* __restrict__ W,
    const float* __restrict__ bias, f16* __restrict__ Y, float outScale)
{
    __shared__ alignas(16) f16 Wl[128][136];   // [o][c], pad 136 keeps 16B alignment
    __shared__ alignas(16) f16 Xl[128][40];    // [n][c-chunk 32], pad 40

    const int tid  = threadIdx.x;
    const int lane = tid & 31;
    const int wave = tid >> 5;
    const int hi   = (lane >= 16) ? 1 : 0;
    const int row  = lane & 15;
    const int b    = blockIdx.x >> 5;
    const int nblk = blockIdx.x & 31;

    // stage W (128x128 f32 -> f16)
    {
        const int o  = tid >> 1;
        const int c0 = (tid & 1) * 64;
        const float* wp = W + o * 128 + c0;
        #pragma unroll
        for (int k = 0; k < 8; ++k) {
            h8 hv;
            #pragma unroll
            for (int j = 0; j < 8; ++j) hv[j] = (f16)wp[k * 8 + j];
            *(h8*)&Wl[o][c0 + k * 8] = hv;
        }
    }

    v8f acc[8];
    v8f zero = {};
    #pragma unroll
    for (int i = 0; i < 8; ++i) acc[i] = zero;

    const int obase = wave * 16;

    for (int cc = 0; cc < 4; ++cc) {
        const int kc = cc * 32;
        __syncthreads();
        // stage X chunk [32 c][128 n] -> Xl[n][c] (transpose + cvt f16)
        {
            const int c  = tid >> 3;
            const int n0 = (tid & 7) * 16;
            const float* xp = X + ((size_t)b * CCH + kc + c) * NPOS + nblk * 128 + n0;
            #pragma unroll
            for (int j = 0; j < 16; ++j) Xl[n0 + j][c] = (f16)xp[j];
            if (cc < 3)  // prefetch next chunk's row into L2/L0
                __builtin_prefetch(xp + 32 * NPOS, 0, 3);
        }
        __syncthreads();

        AF a;
        a.h[0] = *(const h8*)&Wl[obase + row][kc + hi * 8];
        a.h[1] = *(const h8*)&Wl[obase + row][kc + hi * 8 + 16];
        #pragma unroll
        for (int nt = 0; nt < 8; ++nt) {
            AF bb;
            bb.h[0] = *(const h8*)&Xl[nt * 16 + row][hi * 8];
            bb.h[1] = *(const h8*)&Xl[nt * 16 + row][hi * 8 + 16];
            acc[nt] = wmma16(a, bb, acc[nt]);
        }
    }

    float bs[8];
    #pragma unroll
    for (int r = 0; r < 8; ++r) bs[r] = bias[obase + hi * 8 + r];

    #pragma unroll
    for (int nt = 0; nt < 8; ++nt) {
        const int n = nblk * 128 + nt * 16 + row;
        h8 o8;
        #pragma unroll
        for (int r = 0; r < 8; ++r) o8[r] = (f16)((acc[nt][r] + bs[r]) * outScale);
        *(h8*)(Y + ((size_t)b * NPOS + n) * CCH + obase + hi * 8) = o8;
    }
}

// ---------------------------------------------------------------------------
// Kernel 2: flash attention. block = (b, h, 128 queries); 8 waves x 16 queries.
// Streams keys in tiles of 128 with DOUBLE-BUFFERED staging:
//   K tile  : global_load_async_to_lds_b128 (ASYNCcnt path, no VGPR round-trip)
//   V tile  : global_load_b128 -> transposed ds stores (needs dim-major layout)
// Online softmax; attn matrix never touches HBM.
// grid: B * HEADS * (NPOS/128) = 256
// ---------------------------------------------------------------------------
__global__ __launch_bounds__(256) void attn_kernel(
    const f16* __restrict__ Qh, const f16* __restrict__ Kh,
    const f16* __restrict__ Vh, f16* __restrict__ Oh)
{
    __shared__ alignas(16) f16 Kt[2][128][32];    // [buf][key][dim]
    __shared__ alignas(16) f16 Vt[2][32][136];    // [buf][dim][key], pad 136
    __shared__ alignas(16) f16 Pt[8][16][32];     // per-wave P staging [q][key-in-chunk]

    const int tid  = threadIdx.x;
    const int lane = tid & 31;
    const int wave = tid >> 5;
    const int hi   = (lane >= 16) ? 1 : 0;
    const int row  = lane & 15;

    const int qb = blockIdx.x & 31;
    const int h  = (blockIdx.x >> 5) & 3;
    const int b  = blockIdx.x >> 7;

    const int qbase = qb * 128 + wave * 16;

    // stage one 128-key tile into buffer `buf`:
    //   K rows via async global->LDS DMA; V rows via VGPR transpose.
    auto stage = [&](int buf, int key0) {
        #pragma unroll
        for (int it = 0; it < 2; ++it) {
            const int id   = tid + it * 256;
            const int key  = id >> 2;
            const int part = id & 3;
            const size_t g = ((size_t)(b * NPOS + key0 + key)) * CCH + h * HD + part * 8;
            const f16* kp = Kh + g;
            const unsigned ldsK = (unsigned)(uintptr_t)&Kt[buf][key][part * 8];
            asm volatile("global_load_async_to_lds_b128 %0, %1, off"
                         :: "v"(ldsK), "v"(kp) : "memory");
            h8 vv = *(const h8*)(Vh + g);
            #pragma unroll
            for (int j = 0; j < 8; ++j) Vt[buf][part * 8 + j][key] = vv[j];
        }
    };

    // Q A-fragment for this wave (16 q x 32 dims), softmax scale already folded in
    AF aq;
    {
        const f16* qp = Qh + ((size_t)(b * NPOS + qbase + row)) * CCH + h * HD + hi * 8;
        aq.h[0] = *(const h8*)qp;
        aq.h[1] = *(const h8*)(qp + 16);
    }

    float mrun[8], lrun[8];
    #pragma unroll
    for (int r = 0; r < 8; ++r) { mrun[r] = -1e30f; lrun[r] = 0.0f; }
    v8f O0 = {}, O1 = {};

    // prologue: stage tile 0
    stage(0, 0);
    asm volatile("s_wait_asynccnt 0" ::: "memory");
    __syncthreads();

    for (int kt = 0; kt < NPOS / 128; ++kt) {
        const int buf = kt & 1;
        // prefetch next tile into the other buffer (overlaps with WMMAs below);
        // ^buf was last read in iteration kt-1, protected by that iteration's barrier.
        if (kt + 1 < NPOS / 128) stage(buf ^ 1, (kt + 1) * 128);

        #pragma unroll
        for (int cc = 0; cc < 4; ++cc) {
            const int kc = cc * 32;
            // S = Q * K^T : two 16-key tiles, K-dim = head_dim = 32 (one WMMA each)
            AF bk0, bk1;
            bk0.h[0] = *(const h8*)&Kt[buf][kc + row][hi * 8];
            bk0.h[1] = *(const h8*)&Kt[buf][kc + row][hi * 8 + 16];
            bk1.h[0] = *(const h8*)&Kt[buf][kc + 16 + row][hi * 8];
            bk1.h[1] = *(const h8*)&Kt[buf][kc + 16 + row][hi * 8 + 16];
            v8f z = {};
            v8f S0 = wmma16(aq, bk0, z);
            v8f S1 = wmma16(aq, bk1, z);

            // online softmax over the 32-key chunk (rows = queries)
            #pragma unroll
            for (int r = 0; r < 8; ++r) {
                float m = fmaxf(S0[r], S1[r]);
                #pragma unroll
                for (int msk = 1; msk <= 8; msk <<= 1)
                    m = fmaxf(m, __shfl_xor(m, msk));
                const float mnew = fmaxf(mrun[r], m);
                const float corr = __expf(mrun[r] - mnew);
                mrun[r] = mnew;
                const float p0 = __expf(S0[r] - mnew);
                const float p1 = __expf(S1[r] - mnew);
                float s = p0 + p1;
                #pragma unroll
                for (int msk = 1; msk <= 8; msk <<= 1)
                    s += __shfl_xor(s, msk);
                lrun[r] = lrun[r] * corr + s;
                O0[r] *= corr;
                O1[r] *= corr;
                // stage P (C-layout -> LDS) for A-fragment reload
                Pt[wave][hi * 8 + r][row]      = (f16)p0;
                Pt[wave][hi * 8 + r][16 + row] = (f16)p1;
            }
            asm volatile("s_wait_dscnt 0" ::: "memory");

            AF ap;
            ap.h[0] = *(const h8*)&Pt[wave][row][hi * 8];
            ap.h[1] = *(const h8*)&Pt[wave][row][hi * 8 + 16];

            // O += P * V^T : dims 0-15 and 16-31
            AF bv0, bv1;
            bv0.h[0] = *(const h8*)&Vt[buf][row][kc + hi * 8];
            bv0.h[1] = *(const h8*)&Vt[buf][row][kc + hi * 8 + 16];
            bv1.h[0] = *(const h8*)&Vt[buf][16 + row][kc + hi * 8];
            bv1.h[1] = *(const h8*)&Vt[buf][16 + row][kc + hi * 8 + 16];
            O0 = wmma16(ap, bv0, O0);
            O1 = wmma16(ap, bv1, O1);
        }

        // next tile's async copies + V ds stores must land before anyone reads them
        asm volatile("s_wait_asynccnt 0" ::: "memory");
        __syncthreads();
    }

    // normalize and store O (f16, position-major [B][N][C])
    #pragma unroll
    for (int r = 0; r < 8; ++r) {
        const float inv = 1.0f / lrun[r];
        const int   qg  = qbase + hi * 8 + r;
        const size_t bse = ((size_t)(b * NPOS + qg)) * CCH + h * HD;
        Oh[bse + row]      = (f16)(O0[r] * inv);
        Oh[bse + 16 + row] = (f16)(O1[r] * inv);
    }
}

// ---------------------------------------------------------------------------
// Kernel 3: output projection + bias + residual -> R f32 [B][C][N]
// grid: B * (NPOS/128) = 64
// ---------------------------------------------------------------------------
__global__ __launch_bounds__(256) void oproj_kernel(
    const f16* __restrict__ Oh, const float* __restrict__ W,
    const float* __restrict__ bias, const float* __restrict__ Xres,
    float* __restrict__ R)
{
    __shared__ alignas(16) f16 Wl[128][136];

    const int tid  = threadIdx.x;
    const int lane = tid & 31;
    const int wave = tid >> 5;
    const int hi   = (lane >= 16) ? 1 : 0;
    const int row  = lane & 15;
    const int b    = blockIdx.x >> 5;
    const int nblk = blockIdx.x & 31;

    {
        const int o  = tid >> 1;
        const int c0 = (tid & 1) * 64;
        const float* wp = W + o * 128 + c0;
        #pragma unroll
        for (int k = 0; k < 8; ++k) {
            h8 hv;
            #pragma unroll
            for (int j = 0; j < 8; ++j) hv[j] = (f16)wp[k * 8 + j];
            *(h8*)&Wl[o][c0 + k * 8] = hv;
        }
    }
    __syncthreads();

    v8f acc[8];
    v8f zero = {};
    #pragma unroll
    for (int i = 0; i < 8; ++i) acc[i] = zero;

    const int obase = wave * 16;

    for (int cc = 0; cc < 4; ++cc) {
        const int kc = cc * 32;
        AF a;
        a.h[0] = *(const h8*)&Wl[obase + row][kc + hi * 8];
        a.h[1] = *(const h8*)&Wl[obase + row][kc + hi * 8 + 16];
        #pragma unroll
        for (int nt = 0; nt < 8; ++nt) {
            const int n = nblk * 128 + nt * 16 + row;
            const f16* op = Oh + ((size_t)(b * NPOS + n)) * CCH + kc + hi * 8;
            AF bb;
            bb.h[0] = *(const h8*)op;
            bb.h[1] = *(const h8*)(op + 16);
            acc[nt] = wmma16(a, bb, acc[nt]);
        }
    }

    float bs[8];
    #pragma unroll
    for (int r = 0; r < 8; ++r) bs[r] = bias[obase + hi * 8 + r];

    #pragma unroll
    for (int nt = 0; nt < 8; ++nt) {
        const int n = nblk * 128 + nt * 16 + row;
        #pragma unroll
        for (int r = 0; r < 8; ++r) {
            const int o = obase + hi * 8 + r;
            const size_t idx = ((size_t)b * CCH + o) * NPOS + n;
            R[idx] = acc[nt][r] + bs[r] + Xres[idx];
        }
    }
}

// ---------------------------------------------------------------------------
// Kernel 4: InstanceNorm3d (affine=False) over N per (b,c). grid = B*C = 256
// ---------------------------------------------------------------------------
__global__ __launch_bounds__(256) void inorm_kernel(
    const float* __restrict__ R, float* __restrict__ out)
{
    __shared__ float s1[256];
    __shared__ float s2[256];
    const int t = threadIdx.x;
    const float* p = R + (size_t)blockIdx.x * NPOS;

    float v[16];
    float s = 0.0f, sq = 0.0f;
    #pragma unroll
    for (int i = 0; i < 16; ++i) {
        v[i] = p[t + 256 * i];
        s  += v[i];
        sq += v[i] * v[i];
    }
    s1[t] = s; s2[t] = sq;
    __syncthreads();
    for (int off = 128; off > 0; off >>= 1) {
        if (t < off) { s1[t] += s1[t + off]; s2[t] += s2[t + off]; }
        __syncthreads();
    }
    const float mu  = s1[0] * (1.0f / NPOS);
    const float var = s2[0] * (1.0f / NPOS) - mu * mu;
    const float rs  = rsqrtf(var + 1e-5f);

    float* q = out + (size_t)blockIdx.x * NPOS;
    #pragma unroll
    for (int i = 0; i < 16; ++i) q[t + 256 * i] = (v[i] - mu) * rs;
}

// ---------------------------------------------------------------------------
extern "C" void kernel_launch(void* const* d_in, const int* in_sizes, int n_in,
                              void* d_out, int out_size, void* d_ws, size_t ws_size,
                              hipStream_t stream) {
    const float* Xq  = (const float*)d_in[0];
    const float* Xkv = (const float*)d_in[1];
    const float* Wq  = (const float*)d_in[2];
    const float* bq  = (const float*)d_in[3];
    const float* Wk  = (const float*)d_in[4];
    const float* bk  = (const float*)d_in[5];
    const float* Wv  = (const float*)d_in[6];
    const float* bv  = (const float*)d_in[7];
    const float* Wo  = (const float*)d_in[8];
    const float* bo  = (const float*)d_in[9];

    const size_t nElemH = (size_t)2 * NPOS * CCH;  // per f16 buffer
    f16*   Qh = (f16*)d_ws;
    f16*   Kh = Qh + nElemH;
    f16*   Vh = Kh + nElemH;
    f16*   Oh = Vh + nElemH;
    float* R  = (float*)(Oh + nElemH);

    const float scale = 0.17677669529663687f;  // 1/sqrt(head_dim=32), folded into Q

    proj_kernel<<<64, 256, 0, stream>>>(Xq,  Wq, bq, Qh, scale);
    proj_kernel<<<64, 256, 0, stream>>>(Xkv, Wk, bk, Kh, 1.0f);
    proj_kernel<<<64, 256, 0, stream>>>(Xkv, Wv, bv, Vh, 1.0f);
    attn_kernel<<<256, 256, 0, stream>>>(Qh, Kh, Vh, Oh);
    oproj_kernel<<<64, 256, 0, stream>>>(Oh, Wo, bo, Xq, R);
    inorm_kernel<<<256, 256, 0, stream>>>(R, (float*)d_out);
}
